// ExplorationBehavior_32667521253867
// MI455X (gfx1250) — matrix operand order repaired and checked by reading, not compiled, not run on hardware
//
#include <hip/hip_runtime.h>
#include <hip/hip_bf16.h>
#include <math.h>

// ---------------------------------------------------------------------------
// Problem sizes (from the reference)
// ---------------------------------------------------------------------------
#define B_N   32768
#define D_N   1024
#define P_N   100
#define M_N   50
#define H_N   256
#define IN_RAW 1124          // D + P
#define IN_PAD 1152          // padded to a multiple of 32 (WMMA K step)
#define PLACE_PAD 128        // IN_PAD - D_N  (place cells + zero pad)
#define ALPHA 0.1f

typedef __bf16 bf16_t;
typedef __attribute__((ext_vector_type(16))) __bf16 v16bf;
typedef __attribute__((ext_vector_type(8)))  __bf16 v8bf;
typedef __attribute__((ext_vector_type(8)))  float  v8f;

// ---------------------------------------------------------------------------
// Workspace layout (bytes, 256-aligned regions)
// ---------------------------------------------------------------------------
static constexpr size_t OFF_CELL  = 0;                                   // B int32
static constexpr size_t OFF_PLACE = OFF_CELL  + (size_t)B_N * 4;         // B*128 bf16
static constexpr size_t OFF_W1T   = OFF_PLACE + (size_t)B_N * PLACE_PAD * 2;
static constexpr size_t OFF_W2T   = OFF_W1T   + (size_t)H_N * IN_PAD * 2;
static constexpr size_t OFF_WHT   = OFF_W2T   + (size_t)H_N * H_N * 2;
static constexpr size_t OFF_BH    = OFF_WHT   + (size_t)16 * H_N * 2;
static constexpr size_t OFF_H1    = ((OFF_BH + 64 + 255) / 256) * 256;   // B*H bf16
static constexpr size_t OFF_H2    = OFF_H1    + (size_t)B_N * H_N * 2;   // B*H bf16

// ---------------------------------------------------------------------------
// Weight preparation: transpose to N-major bf16 so the WMMA B operand is a
// single contiguous v16bf load per lane (lanes 0-15: K=0..15, lanes 16-31:
// K=16..31 of a 32x16 K-major tile).
// ---------------------------------------------------------------------------
__global__ void prep_w1t_kernel(const float* __restrict__ W1, bf16_t* __restrict__ W1t) {
  int i = blockIdx.x * blockDim.x + threadIdx.x;        // over H_N * IN_PAD
  int n = i / IN_PAD, k = i % IN_PAD;
  W1t[i] = (k < IN_RAW) ? (bf16_t)W1[(size_t)k * H_N + n] : (bf16_t)0.0f;
}

__global__ void prep_w2t_kernel(const float* __restrict__ W2, bf16_t* __restrict__ W2t) {
  int i = blockIdx.x * blockDim.x + threadIdx.x;        // over H_N * H_N
  int n = i / H_N, k = i % H_N;
  W2t[i] = (bf16_t)W2[(size_t)k * H_N + n];
}

// Pack the three heads [Wd(H,9) | Ws(H,1) | Wg(H,1) | zeros] into a 16 x H
// N-major bf16 matrix + a 16-entry f32 bias so all heads are one WMMA N-tile.
__global__ void prep_heads_kernel(const float* __restrict__ Wd, const float* __restrict__ Ws,
                                  const float* __restrict__ Wg, const float* __restrict__ bd,
                                  const float* __restrict__ bs, const float* __restrict__ bg,
                                  bf16_t* __restrict__ Wht, float* __restrict__ bh) {
  int i = blockIdx.x * blockDim.x + threadIdx.x;        // over 16 * H_N
  int n = i / H_N, k = i % H_N;
  float v = 0.0f;
  if (n < 9)       v = Wd[(size_t)k * 9 + n];
  else if (n == 9) v = Ws[k];
  else if (n == 10) v = Wg[k];
  Wht[i] = (bf16_t)v;
  if (i < 16) {
    float bb = 0.0f;
    if (i < 9) bb = bd[i]; else if (i == 9) bb = bs[0]; else if (i == 10) bb = bg[0];
    bh[i] = bb;
  }
}

// ---------------------------------------------------------------------------
// Place cells + novelty + grid cell index.  One thread per batch row.
// ---------------------------------------------------------------------------
__global__ __launch_bounds__(256) void place_novelty_kernel(
    const float* __restrict__ pos, const float* __restrict__ centers,
    const float* __restrict__ widths, const float* __restrict__ visit,
    float* __restrict__ out_place, float* __restrict__ out_nov,
    bf16_t* __restrict__ place_pad, int* __restrict__ cell_idx) {
  __shared__ float sc[P_N * 2];
  __shared__ float sw[P_N];
  for (int i = threadIdx.x; i < P_N * 2; i += blockDim.x) sc[i] = centers[i];
  for (int i = threadIdx.x; i < P_N; i += blockDim.x)     sw[i] = widths[i];
  __syncthreads();

  const int b = blockIdx.x * blockDim.x + threadIdx.x;
  const float px = pos[b * 2 + 0], py = pos[b * 2 + 1];
  int ix = min(max((int)floorf(px * (float)M_N), 0), M_N - 1);
  int iy = min(max((int)floorf(py * (float)M_N), 0), M_N - 1);
  const int cell = ix * M_N + iy;
  cell_idx[b] = cell;
  out_nov[b]  = __expf(-visit[cell] * 0.1f);

  bf16_t* pp = place_pad + (size_t)b * PLACE_PAD;
  float*  po = out_place + (size_t)b * P_N;
#pragma unroll 4
  for (int p = 0; p < P_N; ++p) {
    const float dx = px - sc[p * 2 + 0];
    const float dy = py - sc[p * 2 + 1];
    const float w  = sw[p];
    const float v  = __expf(-(dx * dx + dy * dy) / (2.0f * w * w));
    po[p] = v;
    pp[p] = (bf16_t)v;
  }
#pragma unroll
  for (int p = P_N; p < PLACE_PAD; ++p) pp[p] = (bf16_t)0.0f;  // K-tail pad
}

// ---------------------------------------------------------------------------
// WMMA A-fragment loaders (contiguous K order, matching the B loads):
//   lane<16:  M=lane,    K slots {0..15}
//   lane>=16: M=lane-16, K slots {16..31}
// Both K runs are 8 contiguous elements -> vector loads.
// ---------------------------------------------------------------------------
__device__ __forceinline__ v16bf make_a_frag(const bf16_t* row_ptr, int hi) {
  const v8bf lo = *(const v8bf*)(row_ptr + hi * 16);
  const v8bf hv = *(const v8bf*)(row_ptr + hi * 16 + 8);
  v16bf a;
#pragma unroll
  for (int i = 0; i < 8; ++i) { a[i] = lo[i]; a[8 + i] = hv[i]; }
  return a;
}

// f32 source: two contiguous 8-float runs (4x global_load_b128) converted to
// bf16 in registers (v_cvt_pk_bf16_f32), no LDS round-trip.
__device__ __forceinline__ v16bf make_a_frag_f32(const float* row_ptr, int hi) {
  const float* p0 = row_ptr + hi * 16;
  const float* p1 = row_ptr + hi * 16 + 8;
  v16bf a;
#pragma unroll
  for (int i = 0; i < 8; ++i) { a[i] = (bf16_t)p0[i]; a[8 + i] = (bf16_t)p1[i]; }
  return a;
}

// ---------------------------------------------------------------------------
// GEMM1: relu(x @ W1 + b1) -> h1 (bf16).  x = [brain_state | place | pad].
// Block = 16 rows x 256 cols, 8 waves, each wave owns two 16x16 col tiles
// (one A fragment feeds 2 WMMAs).  A is read directly per lane; the 8-way
// re-read across waves hits L0/L2 (brain_state < 192MB global L2).
// K loop is split so the brain (f32->bf16) and place (bf16) regions are
// branch-free straight-line code.
// ---------------------------------------------------------------------------
__global__ __launch_bounds__(256) void gemm1_kernel(
    const float* __restrict__ brain, const bf16_t* __restrict__ place_pad,
    const bf16_t* __restrict__ W1t, const float* __restrict__ b1,
    bf16_t* __restrict__ h1) {
  const int rowbase = blockIdx.x * 16;
  const int wave = threadIdx.x >> 5;
  const int lane = threadIdx.x & 31;
  const int m  = lane & 15;
  const int hi = lane >> 4;
  const int col0 = wave * 16 + m;
  const int col1 = 128 + wave * 16 + m;

  const float*  arow = brain + (size_t)(rowbase + m) * D_N;
  const bf16_t* prow = place_pad + (size_t)(rowbase + m) * PLACE_PAD;
  const bf16_t* b0p  = W1t + (size_t)col0 * IN_PAD + hi * 16;
  const bf16_t* b1p  = W1t + (size_t)col1 * IN_PAD + hi * 16;

  v8f acc0 = {};
  v8f acc1 = {};

#pragma unroll 2
  for (int kt = 0; kt < D_N / 32; ++kt) {            // brain region: f32 -> bf16
    const v16bf a   = make_a_frag_f32(arow + kt * 32, hi);
    const v16bf bm0 = *(const v16bf*)(b0p + kt * 32);
    const v16bf bm1 = *(const v16bf*)(b1p + kt * 32);
    acc0 = __builtin_amdgcn_wmma_f32_16x16x32_bf16(false, a, false, bm0, (short)0, acc0, false, false);
    acc1 = __builtin_amdgcn_wmma_f32_16x16x32_bf16(false, a, false, bm1, (short)0, acc1, false, false);
  }
#pragma unroll
  for (int kt = 0; kt < PLACE_PAD / 32; ++kt) {      // place region: native bf16
    const v16bf a   = make_a_frag(prow + kt * 32, hi);
    const v16bf bm0 = *(const v16bf*)(b0p + D_N + kt * 32);
    const v16bf bm1 = *(const v16bf*)(b1p + D_N + kt * 32);
    acc0 = __builtin_amdgcn_wmma_f32_16x16x32_bf16(false, a, false, bm0, (short)0, acc0, false, false);
    acc1 = __builtin_amdgcn_wmma_f32_16x16x32_bf16(false, a, false, bm1, (short)0, acc1, false, false);
  }

  const float bias0 = b1[col0];
  const float bias1 = b1[col1];
#pragma unroll
  for (int j = 0; j < 8; ++j) {
    const int row = rowbase + j + hi * 8;            // C/D layout: M = j + 8*hi, N = lane%16
    h1[(size_t)row * H_N + col0] = (bf16_t)fmaxf(acc0[j] + bias0, 0.0f);
    h1[(size_t)row * H_N + col1] = (bf16_t)fmaxf(acc1[j] + bias1, 0.0f);
  }
}

// ---------------------------------------------------------------------------
// GEMM2: relu(h1 @ W2 + b2) -> h2 (bf16).  Same tiling, bf16 source.
// ---------------------------------------------------------------------------
__global__ __launch_bounds__(256) void gemm2_kernel(
    const bf16_t* __restrict__ h1, const bf16_t* __restrict__ W2t,
    const float* __restrict__ b2, bf16_t* __restrict__ h2) {
  const int rowbase = blockIdx.x * 16;
  const int wave = threadIdx.x >> 5;
  const int lane = threadIdx.x & 31;
  const int m  = lane & 15;
  const int hi = lane >> 4;
  const int col0 = wave * 16 + m;
  const int col1 = 128 + wave * 16 + m;

  const bf16_t* arow = h1 + (size_t)(rowbase + m) * H_N;
  const bf16_t* b0p  = W2t + (size_t)col0 * H_N + hi * 16;
  const bf16_t* b1p  = W2t + (size_t)col1 * H_N + hi * 16;

  v8f acc0 = {};
  v8f acc1 = {};

#pragma unroll 2
  for (int kt = 0; kt < H_N / 32; ++kt) {
    const v16bf a   = make_a_frag(arow + kt * 32, hi);
    const v16bf bm0 = *(const v16bf*)(b0p + kt * 32);
    const v16bf bm1 = *(const v16bf*)(b1p + kt * 32);
    acc0 = __builtin_amdgcn_wmma_f32_16x16x32_bf16(false, a, false, bm0, (short)0, acc0, false, false);
    acc1 = __builtin_amdgcn_wmma_f32_16x16x32_bf16(false, a, false, bm1, (short)0, acc1, false, false);
  }

  const float bias0 = b2[col0];
  const float bias1 = b2[col1];
#pragma unroll
  for (int j = 0; j < 8; ++j) {
    const int row = rowbase + j + hi * 8;
    h2[(size_t)row * H_N + col0] = (bf16_t)fmaxf(acc0[j] + bias0, 0.0f);
    h2[(size_t)row * H_N + col1] = (bf16_t)fmaxf(acc1[j] + bias1, 0.0f);
  }
}

// ---------------------------------------------------------------------------
// Heads: one 16-wide WMMA N-tile covers softmax(9)+sigmoid+sigmoid.
// Block = 128 rows, 8 waves, wave w -> rows [w*16, w*16+16).
// ---------------------------------------------------------------------------
__global__ __launch_bounds__(256) void heads_kernel(
    const bf16_t* __restrict__ h2, const bf16_t* __restrict__ Wht,
    const float* __restrict__ bh,
    float* __restrict__ dirp, float* __restrict__ speed, float* __restrict__ expl) {
  __shared__ float sC[8][16][17];                 // per-wave f32 tile, +1 pad vs bank conflicts
  const int wave = threadIdx.x >> 5;
  const int lane = threadIdx.x & 31;
  const int m  = lane & 15;
  const int hi = lane >> 4;
  const int rowbase = blockIdx.x * 128 + wave * 16;

  const bf16_t* arow = h2 + (size_t)(rowbase + m) * H_N;
  const bf16_t* bp   = Wht + (size_t)m * H_N + hi * 16;

  v8f acc = {};
#pragma unroll
  for (int kt = 0; kt < H_N / 32; ++kt) {
    const v16bf a  = make_a_frag(arow + kt * 32, hi);
    const v16bf bm = *(const v16bf*)(bp + kt * 32);
    acc = __builtin_amdgcn_wmma_f32_16x16x32_bf16(false, a, false, bm, (short)0, acc, false, false);
  }

  const float bias = bh[m];
#pragma unroll
  for (int j = 0; j < 8; ++j) sC[wave][j + hi * 8][m] = acc[j] + bias;
  __syncthreads();

  if (lane < 16) {
    const float* r = sC[wave][lane];
    const int gr = rowbase + lane;
    float mx = r[0];
#pragma unroll
    for (int i = 1; i < 9; ++i) mx = fmaxf(mx, r[i]);
    float e[9], s = 0.0f;
#pragma unroll
    for (int i = 0; i < 9; ++i) { e[i] = __expf(r[i] - mx); s += e[i]; }
    const float inv = 1.0f / s;
#pragma unroll
    for (int i = 0; i < 9; ++i) dirp[(size_t)gr * 9 + i] = e[i] * inv;
    speed[gr] = 1.0f / (1.0f + __expf(-r[9]));
    expl[gr]  = 1.0f / (1.0f + __expf(-r[10]));
  }
}

// ---------------------------------------------------------------------------
// Scatter update of the spatial map + visit counts.  new_map was pre-seeded
// with spatial_map; JAX's duplicate-index .at[].add sums deltas computed vs
// the ORIGINAL map, which is exactly what atomicAdd of alpha*(x - old) gives.
// ---------------------------------------------------------------------------
__global__ __launch_bounds__(256) void map_update_kernel(
    const float* __restrict__ brain, const float* __restrict__ smap,
    const int* __restrict__ cell_idx,
    float* __restrict__ new_map, float* __restrict__ new_visit) {
  const int b = blockIdx.x;
  const int cell = cell_idx[b];
  const float* src = brain + (size_t)b * D_N;
  const float* old = smap + (size_t)cell * D_N;
  float* dst = new_map + (size_t)cell * D_N;
  const int d = threadIdx.x * 4;
#pragma unroll
  for (int u = 0; u < 4; ++u)
    atomicAdd(&dst[d + u], ALPHA * (src[d + u] - old[d + u]));
  if (threadIdx.x == 0) atomicAdd(&new_visit[cell], 1.0f);
}

// ---------------------------------------------------------------------------
// Launch
// ---------------------------------------------------------------------------
extern "C" void kernel_launch(void* const* d_in, const int* in_sizes, int n_in,
                              void* d_out, int out_size, void* d_ws, size_t ws_size,
                              hipStream_t stream) {
  (void)in_sizes; (void)n_in; (void)out_size; (void)ws_size;
  const float* brain   = (const float*)d_in[0];
  const float* pos     = (const float*)d_in[1];
  const float* centers = (const float*)d_in[2];
  const float* widths  = (const float*)d_in[3];
  const float* smap    = (const float*)d_in[4];
  const float* visit   = (const float*)d_in[5];
  const float* W1 = (const float*)d_in[6];
  const float* b1 = (const float*)d_in[7];
  const float* W2 = (const float*)d_in[8];
  const float* b2 = (const float*)d_in[9];
  const float* Wd = (const float*)d_in[10];
  const float* bd = (const float*)d_in[11];
  const float* Ws = (const float*)d_in[12];
  const float* bs = (const float*)d_in[13];
  const float* Wg = (const float*)d_in[14];
  const float* bg = (const float*)d_in[15];

  float* out       = (float*)d_out;
  float* out_dir   = out;                                   // B*9
  float* out_speed = out_dir   + (size_t)B_N * 9;           // B
  float* out_expl  = out_speed + (size_t)B_N;               // B
  float* out_nov   = out_expl  + (size_t)B_N;               // B
  float* out_place = out_nov   + (size_t)B_N;               // B*P
  float* out_map   = out_place + (size_t)B_N * P_N;         // M*M*D
  float* out_visit = out_map   + (size_t)M_N * M_N * D_N;   // M*M

  char* ws = (char*)d_ws;
  int*    cell_idx  = (int*)   (ws + OFF_CELL);
  bf16_t* place_pad = (bf16_t*)(ws + OFF_PLACE);
  bf16_t* W1t       = (bf16_t*)(ws + OFF_W1T);
  bf16_t* W2t       = (bf16_t*)(ws + OFF_W2T);
  bf16_t* Wht       = (bf16_t*)(ws + OFF_WHT);
  float*  bh        = (float*) (ws + OFF_BH);
  bf16_t* h1        = (bf16_t*)(ws + OFF_H1);
  bf16_t* h2        = (bf16_t*)(ws + OFF_H2);

  prep_w1t_kernel<<<(H_N * IN_PAD) / 256, 256, 0, stream>>>(W1, W1t);
  prep_w2t_kernel<<<(H_N * H_N) / 256, 256, 0, stream>>>(W2, W2t);
  prep_heads_kernel<<<(16 * H_N) / 256, 256, 0, stream>>>(Wd, Ws, Wg, bd, bs, bg, Wht, bh);

  place_novelty_kernel<<<B_N / 256, 256, 0, stream>>>(
      pos, centers, widths, visit, out_place, out_nov, place_pad, cell_idx);

  gemm1_kernel<<<B_N / 16, 256, 0, stream>>>(brain, place_pad, W1t, b1, h1);
  gemm2_kernel<<<B_N / 16, 256, 0, stream>>>(h1, W2t, b2, h2);
  heads_kernel<<<B_N / 128, 256, 0, stream>>>(h2, Wht, bh, out_dir, out_speed, out_expl);

  hipMemcpyAsync(out_map, smap, sizeof(float) * (size_t)M_N * M_N * D_N,
                 hipMemcpyDeviceToDevice, stream);
  hipMemcpyAsync(out_visit, visit, sizeof(float) * (size_t)M_N * M_N,
                 hipMemcpyDeviceToDevice, stream);
  map_update_kernel<<<B_N, 256, 0, stream>>>(brain, smap, cell_idx, out_map, out_visit);
}
